// MontonicAttention_22557168238907
// MI455X (gfx1250) — matrix operand order repaired and checked by reading, not compiled, not run on hardware
//
#include <hip/hip_runtime.h>
#include <hip/hip_bf16.h>
#include <stdint.h>

typedef __attribute__((ext_vector_type(16))) _Float16 v16h;
typedef __attribute__((ext_vector_type(8)))  float    v8f;
typedef __attribute__((ext_vector_type(4)))  int      v4i;

#define B_   64
#define T_   1024
#define RNN  1024
#define EMB  512
#define AD   128
#define NF   32
#define KS   31
#define NM   8
#define LVMIN (-4.6051702f)      // log(0.1^2)
#define LVMAX ( 7.2093648f)      // log(36.77^2)

#define PMS  132                 // padded LDS row stride for pm tile (banks)

#if defined(__HIP_DEVICE_COMPILE__) && \
    __has_builtin(__builtin_amdgcn_global_load_async_to_lds_b128) && \
    __has_builtin(__builtin_amdgcn_s_wait_asynccnt)
#define HAVE_ASYNC_LDS 1
typedef __attribute__((address_space(1))) v4i* async_gptr_t;  // global (AS1)
typedef __attribute__((address_space(3))) v4i* async_lptr_t;  // LDS (AS3)
#else
#define HAVE_ASYNC_LDS 0
#endif

// ---------------------------------------------------------------------------
// init: zero context output and avg accumulator (harness poisons buffers)
// ---------------------------------------------------------------------------
__global__ __launch_bounds__(256)
void k_init(float* __restrict__ ctx, float* __restrict__ avg_acc) {
  int i = blockIdx.x * 256 + threadIdx.x;
  if (i < B_ * EMB) ctx[i] = 0.0f;
  if (i < B_ * AD)  avg_acc[i] = 0.0f;
}

// ---------------------------------------------------------------------------
// processed_query = h (64x1024) @ Wq^T (Wq is 128x1024) -> pq (64x128)
// one wave per 16x16 output tile, v_wmma_f32_16x16x32_f16, 32 K-chunks
// ---------------------------------------------------------------------------
__global__ __launch_bounds__(32)
void k_query(const float* __restrict__ h, const float* __restrict__ Wq,
             float* __restrict__ pq) {
  const int lane = threadIdx.x & 31;
  const int b0 = blockIdx.x * 16;     // 4 tiles in M (batch)
  const int d0 = blockIdx.y * 16;     // 8 tiles in N (attn dim)
  const int half = (lane >= 16) ? 1 : 0;
  const int mn = lane & 15;

  const float* arow = h  + (size_t)(b0 + mn) * RNN;   // A: M=b, K striped
  const float* brow = Wq + (size_t)(d0 + mn) * RNN;   // B[K][N] = Wq[d][K]

  v8f c = {};
  for (int kc = 0; kc < RNN; kc += 32) {
    v16h a, bb;
#pragma unroll
    for (int e = 0; e < 16; ++e) {
      int ka = kc + e + ((e >= 8) ? 8 : 0) + (half ? 8 : 0);
      int kb = kc + e + ((e >= 8) ? 8 : 0) + (half ? 8 : 0);
      a[e]  = (_Float16)arow[ka];
      bb[e] = (_Float16)brow[kb];
    }
    c = __builtin_amdgcn_wmma_f32_16x16x32_f16(false, a, false, bb,
                                               (short)0, c, false, false);
  }
#pragma unroll
  for (int r = 0; r < 8; ++r) {
    int b = b0 + r + (half ? 8 : 0);
    pq[b * AD + d0 + mn] = c[r];
  }
}

// ---------------------------------------------------------------------------
// conv1d(aw, conv_w) -> loc tile (32f x 16t in LDS); per-tile projection
//   pred[t,d] = tanh(pq[d] + sum_f loc[f,t]*W_loc[d,f] + pm[t,d])
// one v_wmma_f32_16x16x32_f16 per wave (K = NF = 32). The contiguous 8KB pm
// tile is staged into LDS with async global->LDS b128 copies issued at kernel
// entry (overlapped with the conv), completed with s_wait_asynccnt before the
// barrier. Reduce tanh over the tile's 16 t's, atomicAdd into (b,d) sums.
// grid = (B, T/16), block = 256 (8 waves -> 8 d-tiles of 16)
// ---------------------------------------------------------------------------
__global__ __launch_bounds__(256)
void k_loc_tanh(const float* __restrict__ aw,    // (B,2,T)
                const float* __restrict__ cw,    // (32,2,31)
                const float* __restrict__ Wl,    // (128,32)
                const float* __restrict__ pm,    // (B,T,128)
                const float* __restrict__ pq,    // (B,128)
                float* __restrict__ avg_acc) {   // (B,128) running sums
  __shared__ float cw_s[NF * 2 * KS];            // 1984 floats
  __shared__ float aw_s[2][46];                  // 16-t tile + 15 halo each side
  __shared__ float loc_s[NF][17];                // [f][t], padded stride
  __shared__ float pm_s[16 * PMS];               // pm tile, padded rows

  const int b   = blockIdx.x;
  const int t0  = blockIdx.y * 16;
  const int tid = threadIdx.x;

  // ---- stage pm tile (16 rows x 128 floats, contiguous in memory) ----
  const float* pmsrc = pm + ((size_t)b * T_ + t0) * AD;
#if HAVE_ASYNC_LDS
  for (int i = tid; i < 512; i += 256) {           // 512 x 16B chunks
    int row = i >> 5, col = (i & 31) * 4;
    __builtin_amdgcn_global_load_async_to_lds_b128(
        (async_gptr_t)(pmsrc + i * 4),
        (async_lptr_t)(&pm_s[row * PMS + col]),
        0, 0);
  }
#else
  for (int i = tid; i < 512; i += 256) {
    int row = i >> 5, col = (i & 31) * 4;
    *(float4*)(&pm_s[row * PMS + col]) = *(const float4*)(pmsrc + i * 4);
  }
#endif

  for (int i = tid; i < NF * 2 * KS; i += 256) cw_s[i] = cw[i];
  if (tid < 92) {
    int c = tid / 46, i = tid % 46;
    int t = t0 - 15 + i;
    aw_s[c][i] = (t >= 0 && t < T_) ? aw[(b * 2 + c) * T_ + t] : 0.0f;
  }
  __syncthreads();

  // 512 conv outputs, 2 per thread (overlaps the in-flight async copy)
  for (int o = tid; o < NF * 16; o += 256) {
    int f = o >> 4, t = o & 15;
    float s = 0.0f;
#pragma unroll
    for (int k = 0; k < KS; ++k) {
      s += aw_s[0][t + k] * cw_s[(f * 2 + 0) * KS + k];
      s += aw_s[1][t + k] * cw_s[(f * 2 + 1) * KS + k];
    }
    loc_s[f][t] = s;
  }
#if HAVE_ASYNC_LDS
  __builtin_amdgcn_s_wait_asynccnt(0);   // own async copies done before barrier
#endif
  __syncthreads();                        // loc_s + everyone's pm_s visible

  const int wave = tid >> 5;
  const int lane = tid & 31;
  const int half = (lane >= 16) ? 1 : 0;
  const int mn = lane & 15;
  const int d0 = wave * 16;

  v16h a, bb;
#pragma unroll
  for (int e = 0; e < 16; ++e) {
    int ka = e + ((e >= 8) ? 8 : 0) + (half ? 8 : 0); // f index for A[M=t][K=f]
    int kb = e + ((e >= 8) ? 8 : 0) + (half ? 8 : 0); // f index for B[K=f][N=d]
    a[e]  = (_Float16)loc_s[ka][mn];
    bb[e] = (_Float16)Wl[(d0 + mn) * NF + kb];
  }
  v8f c = {};
  c = __builtin_amdgcn_wmma_f32_16x16x32_f16(false, a, false, bb,
                                             (short)0, c, false, false);

  const int d = d0 + mn;
  const float q = pq[b * AD + d];
  float s = 0.0f;
#pragma unroll
  for (int r = 0; r < 8; ++r) {
    float v = c[r] + q + pm_s[(r + (half ? 8 : 0)) * PMS + d];
    s += tanhf(v);
  }
  // lanes L and L+16 hold the same d (other half of the 16 t's): pair-reduce
  s += __shfl_xor(s, 16, 32);
  if (lane < 16) atomicAdd(&avg_acc[b * AD + d], s);
}

// ---------------------------------------------------------------------------
// per-batch stats: avg = acc/T; means = prev + sum softplus(avg@Wm^T);
// std = exp(0.5 * ((LVMAX-LVMIN)*sigmoid(avg@Wlv^T) + LVMIN)); one wave per b
// ---------------------------------------------------------------------------
__global__ __launch_bounds__(32)
void k_stats(const float* __restrict__ avg_acc, const float* __restrict__ Wm,
             const float* __restrict__ Wlv, const float* __restrict__ prev_means,
             float* __restrict__ means, float* __restrict__ stds) {
  const int b = blockIdx.x;
  const int lane = threadIdx.x;
  float a[4];
#pragma unroll
  for (int j = 0; j < 4; ++j)
    a[j] = avg_acc[b * AD + lane + 32 * j] * (1.0f / (float)T_);

  float acc[NM + 1];
#pragma unroll
  for (int m = 0; m < NM; ++m) {
    float s = 0.0f;
#pragma unroll
    for (int j = 0; j < 4; ++j) s += a[j] * Wm[m * AD + lane + 32 * j];
    acc[m] = s;
  }
  {
    float s = 0.0f;
#pragma unroll
    for (int j = 0; j < 4; ++j) s += a[j] * Wlv[lane + 32 * j];
    acc[NM] = s;
  }
#pragma unroll
  for (int m = 0; m <= NM; ++m)
#pragma unroll
    for (int off = 16; off > 0; off >>= 1)
      acc[m] += __shfl_xor(acc[m], off, 32);

  if (lane == 0) {
    float mi = 0.0f;
#pragma unroll
    for (int m = 0; m < NM; ++m) {
      float x = acc[m];
      mi += (x > 20.0f) ? x : log1pf(expf(x));   // softplus
    }
    means[b] = prev_means[b] + mi;
    float lv = (LVMAX - LVMIN) / (1.0f + expf(-acc[NM])) + LVMIN;
    stds[b] = expf(0.5f * lv);
  }
}

// ---------------------------------------------------------------------------
// weights: p = Phi((t+.5-mu)/sd) - Phi((t-.5-mu)/sd), masked; then
// context[b,e] += sum_t w[b,t] * memory[b,t,e]
// Dominant HBM stream: b128 loads (512B/wave/instr), 2 t-rows in flight,
// each thread owns a 4-wide e group; speculative prefetch 8 rows ahead.
// grid = (B, T/128), block = 256; atomicAdd partial contexts.
// ---------------------------------------------------------------------------
__global__ __launch_bounds__(256)
void k_ctx(const float* __restrict__ memory, const uint8_t* __restrict__ mask,
           const float* __restrict__ means, const float* __restrict__ stds,
           float* __restrict__ ctx, float* __restrict__ wout) {
  __shared__ float w_s[128];
  const int b   = blockIdx.x;
  const int t0  = blockIdx.y * 128;
  const int tid = threadIdx.x;
  const float mu  = means[b];
  const float inv = 0.70710678f / stds[b];   // 1/(sd*sqrt(2))

  if (tid < 128) {
    int t = t0 + tid;
    float x = (float)t;
    float p = 0.5f * (erff((x + 0.5f - mu) * inv) - erff((x - 0.5f - mu) * inv));
    float w = mask[b * T_ + t] ? 0.0f : p;
    wout[b * T_ + t] = w;
    w_s[tid] = w;
  }
  __syncthreads();

  const int tpar = tid >> 7;          // which t-parity this thread covers
  const int e4   = (tid & 127) * 4;   // 4-wide e group
  const float* mrow = memory + ((size_t)b * T_ + t0) * EMB;
  float acc0 = 0.f, acc1 = 0.f, acc2 = 0.f, acc3 = 0.f;
  for (int tt = tpar; tt < 128; tt += 2) {
    const float w = w_s[tt];
    const float4 m = *(const float4*)(mrow + (size_t)tt * EMB + e4);
    __builtin_prefetch(mrow + (size_t)(tt + 8) * EMB + e4, 0, 0);
    acc0 += w * m.x; acc1 += w * m.y; acc2 += w * m.z; acc3 += w * m.w;
  }
  atomicAdd(&ctx[b * EMB + e4 + 0], acc0);
  atomicAdd(&ctx[b * EMB + e4 + 1], acc1);
  atomicAdd(&ctx[b * EMB + e4 + 2], acc2);
  atomicAdd(&ctx[b * EMB + e4 + 3], acc3);
}

// ---------------------------------------------------------------------------
extern "C" void kernel_launch(void* const* d_in, const int* in_sizes, int n_in,
                              void* d_out, int out_size, void* d_ws, size_t ws_size,
                              hipStream_t stream) {
  const float*   h    = (const float*)d_in[0];   // (B,RNN)
  const float*   mem  = (const float*)d_in[1];   // (B,T,EMB)
  const float*   pm   = (const float*)d_in[2];   // (B,T,AD)
  const float*   aw   = (const float*)d_in[3];   // (B,2,T)
  const float*   pmn  = (const float*)d_in[4];   // (B,)
  const uint8_t* mask = (const uint8_t*)d_in[5]; // (B,T) bool
  const float*   Wq   = (const float*)d_in[6];   // (AD,RNN)
  const float*   cw   = (const float*)d_in[7];   // (NF,2,KS)
  const float*   Wl   = (const float*)d_in[8];   // (AD,NF)
  const float*   Wm   = (const float*)d_in[9];   // (NM,AD)
  const float*   Wlv  = (const float*)d_in[10];  // (1,AD)

  float* out  = (float*)d_out;
  float* ctx  = out;              // B*EMB floats
  float* wout = out + B_ * EMB;   // B*T floats

  float* ws    = (float*)d_ws;
  float* pq    = ws;              // B*AD
  float* avg   = pq + B_ * AD;    // B*AD
  float* means = avg + B_ * AD;   // B
  float* stds  = means + B_;      // B

  k_init    <<<dim3(128),          dim3(256), 0, stream>>>(ctx, avg);
  k_query   <<<dim3(4, 8),         dim3(32),  0, stream>>>(h, Wq, pq);
  k_loc_tanh<<<dim3(B_, T_ / 16),  dim3(256), 0, stream>>>(aw, cw, Wl, pm, pq, avg);
  k_stats   <<<dim3(B_),           dim3(32),  0, stream>>>(avg, Wm, Wlv, pmn, means, stds);
  k_ctx     <<<dim3(B_, T_ / 128), dim3(256), 0, stream>>>(mem, mask, means, stds, ctx, wout);
}
